// Model_24455543783889
// MI455X (gfx1250) — compile-verified, matrix operand order checked by reading, not run on hardware
//
#include <hip/hip_runtime.h>
#include <hip/hip_bf16.h>
#include <math.h>

// ---------------------------------------------------------------------------
// CDNA5 (gfx1250, wave32) WMMA helpers: D(16x16 f32) += A(16x32 bf16) x B(32x16 bf16)
// A fragment layout (ISA 7.12.2, 16-bit A 16x32): lane L (m = L&15) holds two
// contiguous 8-element K-chunks at k = (L>=16 ? 8 : 0) and k+16.
// B is the weight matrix W[N,K] row-major (we compute X @ W^T), so the B
// fragment (k,n)=W[n,k] loads the same way with n in place of m.
// C/D layout: lane holds (m = v + 8*(lane>=16), n = lane&15) for v in 0..7.
// ---------------------------------------------------------------------------

typedef __attribute__((ext_vector_type(16))) __bf16 v16bf;
typedef __attribute__((ext_vector_type(8)))  float  v8f;
typedef __attribute__((ext_vector_type(4)))  unsigned int v4u;
typedef __attribute__((ext_vector_type(4)))  int v4i;

#if __has_builtin(__builtin_amdgcn_global_load_async_to_lds_b128) && \
    __has_builtin(__builtin_amdgcn_s_wait_asynccnt)
#define HAVE_ASYNC_LDS 1
#else
#define HAVE_ASYNC_LDS 0
#endif

#define AS1 __attribute__((address_space(1)))
#define AS3 __attribute__((address_space(3)))

union Frag {
  v16bf v;
  v4u   q[2];
};

__device__ __forceinline__ void load_frag(Frag& f, const __bf16* __restrict__ p,
                                          int row, int ld, int k0, int lane) {
  const __bf16* base = p + (size_t)(row + (lane & 15)) * ld + k0 + ((lane >> 4) << 3);
  f.q[0] = *(const v4u*)(base);
  f.q[1] = *(const v4u*)(base + 16);
}

// LDS tile: 64 rows x 32 bf16, padded row stride 40 bf16 (80 B) to spread banks.
#define LDS_LD 40
__device__ __forceinline__ void load_frag_lds(Frag& f, const __bf16* tile,
                                              int row16, int lane) {
  const __bf16* base = tile + (row16 + (lane & 15)) * LDS_LD + ((lane >> 4) << 3);
  f.q[0] = *(const v4u*)(base);
  f.q[1] = *(const v4u*)(base + 16);
}

// One 16-byte chunk per thread: global row (caller-computed) -> LDS tile row sr,
// chunk sch (elements). Async path tracked by ASYNCcnt; sync fallback uses DS.
__device__ __forceinline__ void stage16(__bf16* lds_dst, const __bf16* gsrc) {
#if HAVE_ASYNC_LDS
  __builtin_amdgcn_global_load_async_to_lds_b128(
      (AS1 v4i*)(uintptr_t)(const void*)gsrc,
      (AS3 v4i*)(uintptr_t)(void*)lds_dst, 0, 0);
#else
  *(v4u*)lds_dst = *(const v4u*)gsrc;
#endif
}
__device__ __forceinline__ void wait_stage() {
#if HAVE_ASYNC_LDS
  __builtin_amdgcn_s_wait_asynccnt(0);
#endif
}

__device__ __forceinline__ v8f wmma_bf(const Frag& a, const Frag& b, v8f c) {
  return __builtin_amdgcn_wmma_f32_16x16x32_bf16(false, a.v, false, b.v,
                                                 (short)0, c, false, false);
}

__device__ __forceinline__ float sigf(float x) { return 1.0f / (1.0f + __expf(-x)); }

// ---------------------------------------------------------------------------
// fp32 -> bf16 conversion (weights + encoder inputs)
// ---------------------------------------------------------------------------
__global__ void f32_to_bf16(const float* __restrict__ in, __bf16* __restrict__ out, size_t n) {
  size_t i = (size_t)blockIdx.x * blockDim.x + threadIdx.x;
  if (i < n) out[i] = (__bf16)in[i];
}

// ---------------------------------------------------------------------------
// C[M,N](f32) = A[M,K](bf16) @ B[N,K](bf16)^T
// block = 256 threads = 8 waves. Block tile 256(M) x 64(N); wave tile 32 x 64
// (2 A frags, 8 accumulators). B tile (64x32) staged once per block in LDS,
// double-buffered with async global->LDS prefetch one k-tile ahead.
// grid = (M/256, N/64)
// ---------------------------------------------------------------------------
__global__ void gemm_bf16nt(const __bf16* __restrict__ A, const __bf16* __restrict__ B,
                            float* __restrict__ C, int N, int K) {
  __shared__ v4u sBq[2][64 * LDS_LD / 8];
  __bf16* sB0 = (__bf16*)sBq[0];
  __bf16* sB1 = (__bf16*)sBq[1];
  const int tid = threadIdx.x;
  const int lane = tid & 31;
  const int wave = tid >> 5;
  const int m0 = blockIdx.x * 256 + wave * 32;
  const int n0 = blockIdx.y * 64;
  // staging map: thread -> (tile row sr, element chunk sch)
  const int sr = tid >> 2;
  const int sch = (tid & 3) * 8;
  const __bf16* gB = B + (size_t)(n0 + sr) * K + sch;
  __bf16* ldst[2] = { sB0 + sr * LDS_LD + sch, sB1 + sr * LDS_LD + sch };

  v8f acc[2][4] = {};
  const int nkt = K >> 5;
  stage16(ldst[0], gB);                       // preload k-tile 0
  for (int kt = 0; kt < nkt; ++kt) {
    wait_stage();
    __syncthreads();
    if (kt + 1 < nkt) stage16(ldst[(kt + 1) & 1], gB + (kt + 1) * 32);
    Frag a0, a1;
    load_frag(a0, A, m0, K, kt * 32, lane);
    load_frag(a1, A, m0 + 16, K, kt * 32, lane);
    const __bf16* bb = (kt & 1) ? sB1 : sB0;
#pragma unroll
    for (int j = 0; j < 4; ++j) {
      Frag b; load_frag_lds(b, bb, j * 16, lane);
      acc[0][j] = wmma_bf(a0, b, acc[0][j]);
      acc[1][j] = wmma_bf(a1, b, acc[1][j]);
    }
  }
  const int n = lane & 15;
#pragma unroll
  for (int s = 0; s < 2; ++s) {
    const int mbase = m0 + s * 16 + ((lane >> 4) << 3);
#pragma unroll
    for (int j = 0; j < 4; ++j)
#pragma unroll
      for (int v = 0; v < 8; ++v)
        C[(size_t)(mbase + v) * N + n0 + j * 16 + n] = acc[s][j][v];
  }
}

// ---------------------------------------------------------------------------
// Encoder LSTM step (M = 30720 rows, hidden 512, input 64).
// gates = xt@Wih^T + h@Whh^T + b ; c,h update fused in registers.
// block = 256 threads = 8 waves; block tile 256 rows x 16 hidden units
// (wave tile 32 rows, 4 gate accumulators x 2 row sub-tiles).
// B tile = 64 rows (4 gates x 16 units) x 32 k, LDS double-buffered async.
// k-tiles 0..1 come from (xt, Wih), 2..17 from (h, Whh).
// grid (120, 32). h double-buffered across step launches.
// ---------------------------------------------------------------------------
__global__ void enc_lstm_step(const __bf16* __restrict__ xt,
                              const __bf16* __restrict__ Wih,
                              const __bf16* __restrict__ Whh,
                              const float* __restrict__ bih,
                              const float* __restrict__ bhh,
                              const __bf16* __restrict__ h_in,
                              __bf16* __restrict__ h_out,
                              float* __restrict__ c,
                              int first) {
  __shared__ v4u sBq[2][64 * LDS_LD / 8];
  __bf16* sB0 = (__bf16*)sBq[0];
  __bf16* sB1 = (__bf16*)sBq[1];
  const int tid = threadIdx.x;
  const int lane = tid & 31;
  const int wave = tid >> 5;
  const int m0 = blockIdx.x * 256 + wave * 32;
  const int j0 = blockIdx.y * 16;
  // staging map: tile row sr in [0,64): gate = sr>>4, unit = sr&15
  const int sr = tid >> 2;
  const int sch = (tid & 3) * 8;
  const int gRow = (sr >> 4) * 512 + j0 + (sr & 15);
  const __bf16* gWih = Wih + (size_t)gRow * 64 + sch;
  const __bf16* gWhh = Whh + (size_t)gRow * 512 + sch;
  __bf16* ldst[2] = { sB0 + sr * LDS_LD + sch, sB1 + sr * LDS_LD + sch };

  v8f acc[2][4] = {};
  const int nkt = first ? 2 : 18;             // 2 x-tiles + 16 h-tiles
  stage16(ldst[0], gWih);
  for (int kt = 0; kt < nkt; ++kt) {
    wait_stage();
    __syncthreads();
    if (kt + 1 < nkt) {
      const int kn = kt + 1;
      const __bf16* src = (kn < 2) ? (gWih + kn * 32) : (gWhh + (kn - 2) * 32);
      stage16(ldst[kn & 1], src);
    }
    Frag a0, a1;
    if (kt < 2) {
      load_frag(a0, xt, m0, 64, kt * 32, lane);
      load_frag(a1, xt, m0 + 16, 64, kt * 32, lane);
    } else {
      load_frag(a0, h_in, m0, 512, (kt - 2) * 32, lane);
      load_frag(a1, h_in, m0 + 16, 512, (kt - 2) * 32, lane);
    }
    const __bf16* bb = (kt & 1) ? sB1 : sB0;
#pragma unroll
    for (int g = 0; g < 4; ++g) {
      Frag b; load_frag_lds(b, bb, g * 16, lane);
      acc[0][g] = wmma_bf(a0, b, acc[0][g]);
      acc[1][g] = wmma_bf(a1, b, acc[1][g]);
    }
  }
  const int n = lane & 15;
  const int col = j0 + n;
  const float bi  = bih[col]        + bhh[col];
  const float bf_ = bih[512 + col]  + bhh[512 + col];
  const float bg  = bih[1024 + col] + bhh[1024 + col];
  const float bo  = bih[1536 + col] + bhh[1536 + col];
#pragma unroll
  for (int s = 0; s < 2; ++s) {
    const int mbase = m0 + s * 16 + ((lane >> 4) << 3);
#pragma unroll
    for (int v = 0; v < 8; ++v) {
      const size_t idx = (size_t)(mbase + v) * 512 + col;
      float gi = acc[s][0][v] + bi;
      float gf = acc[s][1][v] + bf_;
      float gg = acc[s][2][v] + bg;
      float go = acc[s][3][v] + bo;
      float cp = first ? 0.0f : c[idx];
      float cn = sigf(gf) * cp + sigf(gi) * tanhf(gg);
      float hv = sigf(go) * tanhf(cn);
      c[idx] = cn;
      h_out[idx] = (__bf16)hv;
    }
  }
}

// ctx[r, 0:512] = h ; ctx[r, 512:1024] = c  (bf16, [30720,1024])
__global__ void build_ctx(const __bf16* __restrict__ h, const float* __restrict__ c,
                          __bf16* __restrict__ ctx, size_t rows) {
  size_t i = (size_t)blockIdx.x * blockDim.x + threadIdx.x;
  if (i >= rows * 1024) return;
  size_t r = i >> 10;
  int j = (int)(i & 1023);
  ctx[i] = (j < 512) ? h[r * 512 + j] : (__bf16)c[r * 512 + (j - 512)];
}

// ---------------------------------------------------------------------------
// Bidirectional pyramid-LSTM step. Batch M = 64, hidden 512.
// X* holds precomputed x@Wih^T (f32, [64*T, 2048], row = b*T + t).
// grid (1, 32, 2): y = hidden tile, z = direction. block 128 = 4 waves
// covering the 64 batch rows. Writes hidden into out[(b*T+te)*1024 + dir*512 + j].
// ---------------------------------------------------------------------------
__global__ void bilstm_step(const float* __restrict__ Xf, const float* __restrict__ Xb,
                            const __bf16* __restrict__ Whh_f, const __bf16* __restrict__ Whh_b,
                            const float* __restrict__ bih_f, const float* __restrict__ bhh_f,
                            const float* __restrict__ bih_b, const float* __restrict__ bhh_b,
                            const __bf16* __restrict__ hin_f, __bf16* __restrict__ hout_f,
                            float* __restrict__ c_f,
                            const __bf16* __restrict__ hin_b, __bf16* __restrict__ hout_b,
                            float* __restrict__ c_b,
                            __bf16* __restrict__ out, int T, int t, int first) {
  const int dir = blockIdx.z;
  const int lane = threadIdx.x & 31;
  const int wave = threadIdx.x >> 5;
  const int m0 = wave * 16;
  const int j0 = blockIdx.y * 16;
  const int te = dir ? (T - 1 - t) : t;
  const float*  X   = dir ? Xb : Xf;
  const __bf16* Whh = dir ? Whh_b : Whh_f;
  const float*  bih = dir ? bih_b : bih_f;
  const float*  bhh = dir ? bhh_b : bhh_f;
  const __bf16* hin = dir ? hin_b : hin_f;
  __bf16* hout = dir ? hout_b : hout_f;
  float*  c    = dir ? c_b : c_f;
  v8f acc[4] = {};
  if (!first) {
    for (int kt = 0; kt < 16; ++kt) {
      Frag a; load_frag(a, hin, m0, 512, kt * 32, lane);
#pragma unroll
      for (int g = 0; g < 4; ++g) {
        Frag b; load_frag(b, Whh, g * 512 + j0, 512, kt * 32, lane);
        acc[g] = wmma_bf(a, b, acc[g]);
      }
    }
  }
  const int n = lane & 15;
  const int mbase = m0 + ((lane >> 4) << 3);
  const int col = j0 + n;
  const float bi  = bih[col]        + bhh[col];
  const float bf_ = bih[512 + col]  + bhh[512 + col];
  const float bg  = bih[1024 + col] + bhh[1024 + col];
  const float bo  = bih[1536 + col] + bhh[1536 + col];
#pragma unroll
  for (int v = 0; v < 8; ++v) {
    const int mb = mbase + v;                       // batch index
    const size_t xrow = ((size_t)mb * T + te) * 2048;
    float gi = acc[0][v] + X[xrow + col]        + bi;
    float gf = acc[1][v] + X[xrow + 512 + col]  + bf_;
    float gg = acc[2][v] + X[xrow + 1024 + col] + bg;
    float go = acc[3][v] + X[xrow + 1536 + col] + bo;
    const size_t hidx = (size_t)mb * 512 + col;
    float cp = first ? 0.0f : c[hidx];
    float cn = sigf(gf) * cp + sigf(gi) * tanhf(gg);
    float hv = sigf(go) * tanhf(cn);
    c[hidx] = cn;
    hout[hidx] = (__bf16)hv;
    out[((size_t)mb * T + te) * 1024 + dir * 512 + col] = (__bf16)hv;
  }
}

// hid[b, 0:512] = h_fwd_final[b]; hid[b, 512:1024] = h_bwd_final[b]
__global__ void build_hid(const __bf16* __restrict__ hf, const __bf16* __restrict__ hb,
                          float* __restrict__ hid) {
  int i = blockIdx.x * blockDim.x + threadIdx.x;
  if (i >= 64 * 1024) return;
  int b = i >> 10, j = i & 1023;
  hid[i] = (float)((j < 512) ? hf[b * 512 + j] : hb[b * 512 + (j - 512)]);
}

// ---------------------------------------------------------------------------
// Decoder step, part 1: attention scores + softmax + context + comb(ReLU).
// One block per batch row (grid 64, block 128). M=64 is too skinny for WMMA.
// ---------------------------------------------------------------------------
__global__ void dec_attn(const float* __restrict__ inp, const float* __restrict__ hd,
                         const __bf16* __restrict__ enc,
                         const float* __restrict__ attn_W, const float* __restrict__ attn_b,
                         const float* __restrict__ comb_W, const float* __restrict__ comb_b,
                         float* __restrict__ comb) {
  __shared__ float ain[1032];
  __shared__ float wsm[128];
  __shared__ float cin[1032];
  const int b = blockIdx.x;
  const int tid = threadIdx.x;
  if (tid < 8) ain[tid] = inp[b * 8 + tid];
  for (int k = tid; k < 1024; k += 128) ain[8 + k] = hd[b * 1024 + k];
  __syncthreads();
  if (tid < 120) {
    const float* wr = attn_W + (size_t)tid * 1032;
    float s = attn_b[tid];
    for (int k = 0; k < 1032; ++k) s += ain[k] * wr[k];
    wsm[tid] = s;
  }
  __syncthreads();
  if (tid == 0) {
    float m = wsm[0];
    for (int t2 = 1; t2 < 120; ++t2) m = fmaxf(m, wsm[t2]);
    float ssum = 0.0f;
    for (int t2 = 0; t2 < 120; ++t2) { float e = __expf(wsm[t2] - m); wsm[t2] = e; ssum += e; }
    float inv = 1.0f / ssum;
    for (int t2 = 0; t2 < 120; ++t2) wsm[t2] *= inv;
  }
  __syncthreads();
  if (tid < 8) cin[tid] = ain[tid];
  for (int d = tid; d < 1024; d += 128) {
    float s = 0.0f;
    const __bf16* er = enc + (size_t)b * 120 * 1024 + d;
    for (int t2 = 0; t2 < 120; ++t2) s += wsm[t2] * (float)er[(size_t)t2 * 1024];
    cin[8 + d] = s;
  }
  __syncthreads();
  for (int o = tid; o < 512; o += 128) {
    const float* wr = comb_W + (size_t)o * 1032;
    float s = comb_b[o];
    for (int k = 0; k < 1032; ++k) s += cin[k] * wr[k];
    comb[b * 512 + o] = fmaxf(s, 0.0f);
  }
}

// ---------------------------------------------------------------------------
// Decoder step, part 2: GRU cell (hidden 1024) + output proj + log_softmax.
// One block per batch row (grid 64, block 256). Updates hd in place, feeds
// log-probs back into inp for the next step and writes d_out[b,step,:].
// ---------------------------------------------------------------------------
__global__ void dec_gru_out(const float* __restrict__ comb, float* __restrict__ hd,
                            const float* __restrict__ Wih, const float* __restrict__ Whh,
                            const float* __restrict__ bih, const float* __restrict__ bhh,
                            const float* __restrict__ outW, const float* __restrict__ outb,
                            float* __restrict__ dout, float* __restrict__ inp, int step) {
  __shared__ float sc[512];
  __shared__ float sh[1024];
  __shared__ float shn[1024];
  __shared__ float sl[8];
  __shared__ float red[2];
  const int b = blockIdx.x;
  const int tid = threadIdx.x;
  for (int k = tid; k < 512; k += 256) sc[k] = comb[b * 512 + k];
  for (int k = tid; k < 1024; k += 256) sh[k] = hd[b * 1024 + k];
  __syncthreads();
  for (int j = tid; j < 1024; j += 256) {
    float ir = bih[j], iz = bih[1024 + j], in_ = bih[2048 + j];
    const float* wr = Wih + (size_t)j * 512;
    const float* wz = Wih + (size_t)(1024 + j) * 512;
    const float* wn = Wih + (size_t)(2048 + j) * 512;
    for (int k = 0; k < 512; ++k) {
      float x = sc[k];
      ir += x * wr[k]; iz += x * wz[k]; in_ += x * wn[k];
    }
    float hr = bhh[j], hz = bhh[1024 + j], hn = bhh[2048 + j];
    const float* ur = Whh + (size_t)j * 1024;
    const float* uz = Whh + (size_t)(1024 + j) * 1024;
    const float* un = Whh + (size_t)(2048 + j) * 1024;
    for (int k = 0; k < 1024; ++k) {
      float h = sh[k];
      hr += h * ur[k]; hz += h * uz[k]; hn += h * un[k];
    }
    float r = sigf(ir + hr);
    float z = sigf(iz + hz);
    float nn = tanhf(in_ + r * hn);
    shn[j] = (1.0f - z) * nn + z * sh[j];
  }
  __syncthreads();
  for (int k = tid; k < 1024; k += 256) hd[b * 1024 + k] = shn[k];
  if (tid < 8) {
    const float* wr = outW + (size_t)tid * 1024;
    float s = outb[tid];
    for (int k = 0; k < 1024; ++k) s += shn[k] * wr[k];
    sl[tid] = s;
  }
  __syncthreads();
  if (tid == 0) {
    float m = sl[0];
    for (int l = 1; l < 8; ++l) m = fmaxf(m, sl[l]);
    float ssum = 0.0f;
    for (int l = 0; l < 8; ++l) ssum += __expf(sl[l] - m);
    red[0] = m; red[1] = __logf(ssum);
  }
  __syncthreads();
  if (tid < 8) {
    float val = sl[tid] - red[0] - red[1];
    dout[((size_t)b * 32 + step) * 8 + tid] = val;
    inp[b * 8 + tid] = val;
  }
}

// ---------------------------------------------------------------------------
// Host orchestration
// ---------------------------------------------------------------------------
extern "C" void kernel_launch(void* const* d_in, const int* in_sizes, int n_in,
                              void* d_out, int out_size, void* d_ws, size_t ws_size,
                              hipStream_t stream) {
  (void)in_sizes; (void)n_in; (void)out_size; (void)ws_size;
  const float* xs     = (const float*)d_in[0];
  const float* encWih = (const float*)d_in[1];
  const float* encWhh = (const float*)d_in[2];
  const float* encbih = (const float*)d_in[3];
  const float* encbhh = (const float*)d_in[4];
  const float* pWih[4] = { (const float*)d_in[5],  (const float*)d_in[9],
                           (const float*)d_in[13], (const float*)d_in[17] };
  const float* pWhh[4] = { (const float*)d_in[6],  (const float*)d_in[10],
                           (const float*)d_in[14], (const float*)d_in[18] };
  const float* pbih[4] = { (const float*)d_in[7],  (const float*)d_in[11],
                           (const float*)d_in[15], (const float*)d_in[19] };
  const float* pbhh[4] = { (const float*)d_in[8],  (const float*)d_in[12],
                           (const float*)d_in[16], (const float*)d_in[20] };
  const float* attnW = (const float*)d_in[21];
  const float* attnb = (const float*)d_in[22];
  const float* combW = (const float*)d_in[23];
  const float* combb = (const float*)d_in[24];
  const float* gruWih = (const float*)d_in[25];
  const float* gruWhh = (const float*)d_in[26];
  const float* grubih = (const float*)d_in[27];
  const float* grubhh = (const float*)d_in[28];
  const float* outW = (const float*)d_in[29];
  const float* outb = (const float*)d_in[30];
  float* dout = (float*)d_out;

  char* ws = (char*)d_ws;
  size_t off = 0;
  auto alloc = [&](size_t bytes) -> void* {
    void* p = ws + off;
    off += (bytes + 255) & ~(size_t)255;
    return p;
  };
  __bf16* xs_bf   = (__bf16*)alloc((size_t)4 * 30720 * 64 * 2);
  __bf16* eWih_bf = (__bf16*)alloc((size_t)2048 * 64 * 2);
  __bf16* eWhh_bf = (__bf16*)alloc((size_t)2048 * 512 * 2);
  __bf16* pWih_bf[4]; for (int i = 0; i < 4; ++i) pWih_bf[i] = (__bf16*)alloc((size_t)2048 * 2048 * 2);
  __bf16* pWhh_bf[4]; for (int i = 0; i < 4; ++i) pWhh_bf[i] = (__bf16*)alloc((size_t)2048 * 512 * 2);
  __bf16* hEnc[2];    for (int i = 0; i < 2; ++i) hEnc[i] = (__bf16*)alloc((size_t)30720 * 512 * 2);
  float*  cEnc = (float*) alloc((size_t)30720 * 512 * 4);
  __bf16* ctx  = (__bf16*)alloc((size_t)30720 * 1024 * 2);
  float*  Xf   = (float*) alloc((size_t)15360 * 2048 * 4);   // reused by layer 2
  float*  Xb   = (float*) alloc((size_t)15360 * 2048 * 4);   // reused by layer 2
  __bf16* out1 = (__bf16*)alloc((size_t)15360 * 1024 * 2);
  __bf16* enc2 = (__bf16*)alloc((size_t)7680 * 1024 * 2);
  __bf16* hL[2][2][2];                                       // [layer][dir][buf]
  for (int l = 0; l < 2; ++l)
    for (int d = 0; d < 2; ++d)
      for (int q = 0; q < 2; ++q)
        hL[l][d][q] = (__bf16*)alloc((size_t)64 * 512 * 2);
  float* cL[2][2];
  for (int l = 0; l < 2; ++l)
    for (int d = 0; d < 2; ++d)
      cL[l][d] = (float*)alloc((size_t)64 * 512 * 4);
  float* hid  = (float*)alloc((size_t)64 * 1024 * 4);
  float* inp  = (float*)alloc((size_t)64 * 8 * 4);
  float* comb = (float*)alloc((size_t)64 * 512 * 4);

  auto cvt = [&](const float* s, __bf16* dst, size_t n) {
    f32_to_bf16<<<dim3((unsigned)((n + 255) / 256)), dim3(256), 0, stream>>>(s, dst, n);
  };
  cvt(xs, xs_bf, (size_t)4 * 30720 * 64);
  cvt(encWih, eWih_bf, (size_t)2048 * 64);
  cvt(encWhh, eWhh_bf, (size_t)2048 * 512);
  for (int i = 0; i < 4; ++i) {
    cvt(pWih[i], pWih_bf[i], (size_t)2048 * 2048);
    cvt(pWhh[i], pWhh_bf[i], (size_t)2048 * 512);
  }

  // ---- encoder LSTM: 4 steps over 30720 rows ----
  for (int t = 0; t < 4; ++t) {
    enc_lstm_step<<<dim3(120, 32), dim3(256), 0, stream>>>(
        xs_bf + (size_t)t * 30720 * 64, eWih_bf, eWhh_bf, encbih, encbhh,
        hEnc[t & 1], hEnc[(t + 1) & 1], cEnc, t == 0);
  }
  {
    size_t n = (size_t)30720 * 1024;
    build_ctx<<<dim3((unsigned)((n + 255) / 256)), dim3(256), 0, stream>>>(hEnc[0], cEnc, ctx, 30720);
  }

  // ---- pyramid layer 1: precompute x@Wih^T, then 240 recurrent steps ----
  gemm_bf16nt<<<dim3(60, 32), dim3(256), 0, stream>>>(ctx, pWih_bf[0], Xf, 2048, 2048);
  gemm_bf16nt<<<dim3(60, 32), dim3(256), 0, stream>>>(ctx, pWih_bf[1], Xb, 2048, 2048);
  for (int t = 0; t < 240; ++t) {
    bilstm_step<<<dim3(1, 32, 2), dim3(128), 0, stream>>>(
        Xf, Xb, pWhh_bf[0], pWhh_bf[1], pbih[0], pbhh[0], pbih[1], pbhh[1],
        hL[0][0][t & 1], hL[0][0][(t + 1) & 1], cL[0][0],
        hL[0][1][t & 1], hL[0][1][(t + 1) & 1], cL[0][1],
        out1, 240, t, t == 0);
  }

  // ---- pyramid layer 2: 7680 x 2048 inputs, 120 steps ----
  gemm_bf16nt<<<dim3(30, 32), dim3(256), 0, stream>>>(out1, pWih_bf[2], Xf, 2048, 2048);
  gemm_bf16nt<<<dim3(30, 32), dim3(256), 0, stream>>>(out1, pWih_bf[3], Xb, 2048, 2048);
  for (int t = 0; t < 120; ++t) {
    bilstm_step<<<dim3(1, 32, 2), dim3(128), 0, stream>>>(
        Xf, Xb, pWhh_bf[2], pWhh_bf[3], pbih[2], pbhh[2], pbih[3], pbhh[3],
        hL[1][0][t & 1], hL[1][0][(t + 1) & 1], cL[1][0],
        hL[1][1][t & 1], hL[1][1][(t + 1) & 1], cL[1][1],
        enc2, 120, t, t == 0);
  }
  // final hidden after even T lands in buffer 0 for both directions
  build_hid<<<dim3((64 * 1024 + 255) / 256), dim3(256), 0, stream>>>(hL[1][0][0], hL[1][1][0], hid);

  // ---- attention decoder: 32 steps ----
  (void)hipMemsetAsync(inp, 0, 64 * 8 * sizeof(float), stream);
  for (int s = 0; s < 32; ++s) {
    dec_attn<<<dim3(64), dim3(128), 0, stream>>>(inp, hid, enc2, attnW, attnb, combW, combb, comb);
    dec_gru_out<<<dim3(64), dim3(256), 0, stream>>>(comb, hid, gruWih, gruWhh, grubih, grubhh,
                                                    outW, outb, dout, inp, s);
  }
}
